// TGCN_32280974197456
// MI455X (gfx1250) — matrix operand order repaired and checked by reading, not compiled, not run on hardware
//
#include <hip/hip_runtime.h>

// ---------------------------------------------------------------------------
// TGCN (GCN + GRU + FC) for gfx1250 / MI455X.
// GEMMs: v_wmma_f32_16x16x32_bf16, fp32 accumulation.
// Staging: global_load_async_to_lds_b128 (ASYNCcnt) with double-buffered LDS.
// Weights converted once to bf16 (~53MB, L2-resident in 192MB L2).
// ---------------------------------------------------------------------------

typedef __attribute__((ext_vector_type(16))) __bf16   v16bf;
typedef __attribute__((ext_vector_type(8)))  float    v8f;
typedef __attribute__((ext_vector_type(4)))  unsigned u32x4;

union Frag { v16bf v; u32x4 q[2]; };

constexpr int Bb   = 64;          // batch
constexpr int Ll   = 64;          // seq len
constexpr int Nn   = 64;          // nodes
constexpr int Ff   = 8;           // node features
constexpr int Hh   = 32;          // hidden channels
constexpr int NH   = 2048;        // N*H (GRU width)
constexpr int G3   = 6144;        // 3*NH
constexpr int Ee   = 2048;        // edges
constexpr int OUTC = 12 * 64;     // PRE_LEN * N = 768
constexpr int KSTR = 40;          // padded LDS row stride (ushorts) for 32-wide k chunk

__device__ inline unsigned short f2bf(float f) {
  unsigned u = __builtin_bit_cast(unsigned, f);
  return (unsigned short)((u + 0x7fffu + ((u >> 16) & 1u)) >> 16);
}

__device__ inline v8f wmma_bf16(const Frag& a, const Frag& b, v8f c) {
  // D = A(16x32 bf16) x B(32x16 bf16) + C(16x16 f32)
  return __builtin_amdgcn_wmma_f32_16x16x32_bf16(false, a.v, false, b.v,
                                                 (short)0, c, false, false);
}

// Raw 32-bit LDS byte offset of a generic shared pointer (aperture rule:
// the offset is the low 32 bits of the address).
__device__ inline unsigned lds_off(const void* p) {
  return (unsigned)(unsigned long long)p;
}

// Async DMA one 16B chunk per lane: LDS[lds] = MEM[saddr + voff] (ASYNCcnt).
__device__ inline void async_copy_b128(unsigned lds, const void* gbase, unsigned goff) {
  asm volatile("global_load_async_to_lds_b128 %0, %1, %2"
               :: "v"(lds), "v"(goff), "s"(gbase)
               : "memory");
}

// A fragment (16x32, 16-bit): lanes 0-15 hold K {0..7, 16..23}, lanes 16-31 {8..15, 24..31}.
__device__ inline Frag load_fragA(const unsigned short* s, int row_base, int lane) {
  int r  = row_base + (lane & 15);
  int kb = (lane >> 4) << 3;
  Frag f;
  f.q[0] = *(const u32x4*)(s + r * KSTR + kb);
  f.q[1] = *(const u32x4*)(s + r * KSTR + kb + 16);
  return f;
}

// B fragment (32x16, 16-bit): lanes 0-15 hold K {0..7, 16..23} of column n, lanes 16-31 {8..15, 24..31}.
__device__ inline Frag load_fragB(const unsigned short* s, int row_base, int lane) {
  int r  = row_base + (lane & 15);
  int kb = (lane >> 4) << 3;
  Frag f;
  f.q[0] = *(const u32x4*)(s + r * KSTR + kb);
  f.q[1] = *(const u32x4*)(s + r * KSTR + kb + 16);
  return f;
}

// ---------------------------------------------------------------------------
// Small prep kernels
// ---------------------------------------------------------------------------
__global__ void k_dinv(const long long* __restrict__ ei, float* __restrict__ dinv) {
  int n = threadIdx.x;
  if (n < Nn) {
    float deg = 1.f;                      // self loop
    for (int e = 0; e < Ee; ++e) deg += (ei[Ee + e] == (long long)n) ? 1.f : 0.f;
    dinv[n] = rsqrtf(deg);
  }
}

__global__ void k_f2bf(const float* __restrict__ src, unsigned short* __restrict__ dst, int n) {
  for (int i = blockIdx.x * blockDim.x + threadIdx.x; i < n; i += gridDim.x * blockDim.x)
    dst[i] = f2bf(src[i]);
}

__global__ void k_zero_h(float* __restrict__ hf, unsigned short* __restrict__ hb, int n) {
  for (int i = blockIdx.x * blockDim.x + threadIdx.x; i < n; i += gridDim.x * blockDim.x) {
    hf[i] = 0.f;
    hb[i] = 0;
  }
}

// ---------------------------------------------------------------------------
// GCN + ReLU -> h_seq (bf16), one block per (b, l)
// ---------------------------------------------------------------------------
__global__ void __launch_bounds__(256) k_gcn(
    const float* __restrict__ x, const long long* __restrict__ ei,
    const float* __restrict__ Wg, const float* __restrict__ bg,
    const float* __restrict__ dinv, unsigned short* __restrict__ hseq)
{
  __shared__ float sW[Ff * Hh];
  __shared__ float sb[Hh];
  __shared__ float sd[Nn];
  __shared__ float sx[Nn * Ff];
  __shared__ float sh[Nn * Hh];
  __shared__ float sa[Nn * Hh];
  const int tid = threadIdx.x;
  const int b = blockIdx.x, l = blockIdx.y;

  sW[tid] = Wg[tid];                       // 256 == Ff*Hh
  if (tid < Hh) sb[tid] = bg[tid];
  if (tid < Nn) sd[tid] = dinv[tid];
  const float* xb = x + ((size_t)b * Ll + l) * Nn * Ff;
  sx[tid]       = xb[tid];
  sx[tid + 256] = xb[tid + 256];
  __syncthreads();

#pragma unroll
  for (int i = 0; i < 8; ++i) {            // 2048 outputs / 256 threads
    int o = tid + i * 256, n = o >> 5, h = o & 31;
    float s = 0.f;
#pragma unroll
    for (int f = 0; f < Ff; ++f) s += sx[n * Ff + f] * sW[f * Hh + h];
    sh[o] = s;
  }
  __syncthreads();

  const float* src = sh;
  if (b == 0) {                            // edge ids < N == 64 -> only batch-0 rows aggregate
#pragma unroll
    for (int i = 0; i < 8; ++i) {
      int o = tid + i * 256, n = o >> 5;
      sa[o] = sd[n] * sd[n] * sh[o];       // self-loop term
    }
    __syncthreads();
    for (int e = tid; e < Ee; e += 256) {
      int es = (int)ei[e], ed = (int)ei[Ee + e];
      float wn = sd[es] * sd[ed];
#pragma unroll
      for (int h = 0; h < Hh; ++h)
        atomicAdd(&sa[ed * Hh + h], wn * sh[es * Hh + h]);   // ds_add_f32
    }
    __syncthreads();
    src = sa;
  }

  unsigned short* dst = hseq + ((size_t)l * Bb + b) * NH;    // row = l*B + b
#pragma unroll
  for (int i = 0; i < 8; ++i) {
    int o = tid + i * 256, h = o & 31;
    dst[o] = f2bf(fmaxf(src[o] + sb[h], 0.f));
  }
}

// ---------------------------------------------------------------------------
// gates_x = h_seq @ W_ih^T + b_ih   ([4096,2048] x [2048,6144] -> f32)
// 128x128 tile/block, 8 waves, wave = 2x4 16x16 tiles.
// Double-buffered async-LDS staging: 4 b128 copies per thread per k-chunk.
// ---------------------------------------------------------------------------
__global__ void __launch_bounds__(256) k_gatesx(
    const unsigned short* __restrict__ hseq, const unsigned short* __restrict__ wih,
    const float* __restrict__ bih, float* __restrict__ gx)
{
  __shared__ __align__(16) unsigned short As[2][128 * KSTR];
  __shared__ __align__(16) unsigned short Bs[2][128 * KSTR];
  const int tid = threadIdx.x, lane = tid & 31, w = tid >> 5;
  const int m0 = blockIdx.y * 128, n0 = blockIdx.x * 128;
  const int wm = (w & 3) * 2, wn = (w >> 2) * 4;
  const unsigned short* gA = hseq + (size_t)m0 * NH;
  const unsigned short* gB = wih  + (size_t)n0 * NH;
  const unsigned ldsA = lds_off(&As[0][0]);
  const unsigned ldsB = lds_off(&Bs[0][0]);
  constexpr unsigned BUFB = 128 * KSTR * 2;   // bytes per LDS buffer
  v8f acc[2][4] = {};

  auto issue = [&](int buf, int k0) {
#pragma unroll
    for (int i = 0; i < 2; ++i) {
      int idx = tid + i * 256;
      int r = idx >> 2, c = (idx & 3) << 3;
      unsigned lo = (unsigned)buf * BUFB + (unsigned)(r * KSTR + c) * 2u;
      unsigned go = (unsigned)(r * NH + k0 + c) * 2u;
      async_copy_b128(ldsA + lo, gA, go);
      async_copy_b128(ldsB + lo, gB, go);
    }
  };

  issue(0, 0);
  for (int k0 = 0; k0 < NH; k0 += 32) {
    const int cur = (k0 >> 5) & 1;
    if (k0 + 32 < NH) {
      issue(cur ^ 1, k0 + 32);
      asm volatile("s_wait_asynccnt 4" ::: "memory");  // current chunk landed
    } else {
      asm volatile("s_wait_asynccnt 0" ::: "memory");
    }
    __syncthreads();
    Frag a0 = load_fragA(As[cur], (wm + 0) * 16, lane);
    Frag a1 = load_fragA(As[cur], (wm + 1) * 16, lane);
#pragma unroll
    for (int j = 0; j < 4; ++j) {
      Frag bf = load_fragB(Bs[cur], (wn + j) * 16, lane);
      acc[0][j] = wmma_bf16(a0, bf, acc[0][j]);
      acc[1][j] = wmma_bf16(a1, bf, acc[1][j]);
    }
    __syncthreads();                       // buffer 'cur' reusable next+1 iter
  }

  const int rofs = (lane >> 4) << 3, cofs = lane & 15;
#pragma unroll
  for (int im = 0; im < 2; ++im)
#pragma unroll
    for (int j = 0; j < 4; ++j)
#pragma unroll
      for (int v = 0; v < 8; ++v) {
        int m = m0 + (wm + im) * 16 + v + rofs;
        int c = n0 + (wn + j) * 16 + cofs;
        gx[(size_t)m * G3 + c] = acc[im][j][v] + bih[c];
      }
}

// ---------------------------------------------------------------------------
// One GRU step: gh = h @ W_hh^T for 3 gates over this block's 64 columns,
// fused sigmoid/tanh update. 32 blocks/step, async double-buffered staging.
// ---------------------------------------------------------------------------
__global__ void __launch_bounds__(256) k_gru_step(
    int t,
    const unsigned short* __restrict__ hb_in, const float* __restrict__ hf_in,
    const unsigned short* __restrict__ whh,   const float* __restrict__ bhh,
    const float* __restrict__ gx,
    float* __restrict__ hf_out, unsigned short* __restrict__ hb_out)
{
  __shared__ __align__(16) unsigned short As[2][64 * KSTR];
  __shared__ __align__(16) unsigned short Bs[2][192 * KSTR];
  const int tid = threadIdx.x, lane = tid & 31, w = tid >> 5;
  const int jb = blockIdx.x;               // 64-column slice of NH
  const int mi = w & 3, nh = w >> 2;
  const unsigned ldsA = lds_off(&As[0][0]);
  const unsigned ldsB = lds_off(&Bs[0][0]);
  constexpr unsigned ABUF = 64 * KSTR * 2;
  constexpr unsigned BBUF = 192 * KSTR * 2;
  v8f acc[3][2] = {};                      // r, z, n panels

  auto issue = [&](int buf, int k0) {
    {
      int r = tid >> 2, c = (tid & 3) << 3;
      async_copy_b128(ldsA + (unsigned)buf * ABUF + (unsigned)(r * KSTR + c) * 2u,
                      hb_in, (unsigned)(r * NH + k0 + c) * 2u);
    }
#pragma unroll
    for (int i = 0; i < 3; ++i) {
      int idx = tid + i * 256;
      int r = idx >> 2, c = (idx & 3) << 3;
      int grow = (r >> 6) * NH + jb * 64 + (r & 63);   // gate g = r/64
      async_copy_b128(ldsB + (unsigned)buf * BBUF + (unsigned)(r * KSTR + c) * 2u,
                      whh, (unsigned)(grow * NH + k0 + c) * 2u);
    }
  };

  issue(0, 0);
  for (int k0 = 0; k0 < NH; k0 += 32) {
    const int cur = (k0 >> 5) & 1;
    if (k0 + 32 < NH) {
      issue(cur ^ 1, k0 + 32);
      asm volatile("s_wait_asynccnt 4" ::: "memory");
    } else {
      asm volatile("s_wait_asynccnt 0" ::: "memory");
    }
    __syncthreads();
    Frag a = load_fragA(As[cur], mi * 16, lane);
#pragma unroll
    for (int g = 0; g < 3; ++g)
#pragma unroll
      for (int nj = 0; nj < 2; ++nj) {
        Frag bf = load_fragB(Bs[cur], g * 64 + nh * 32 + nj * 16, lane);
        acc[g][nj] = wmma_bf16(a, bf, acc[g][nj]);
      }
    __syncthreads();
  }

  const int rofs = (lane >> 4) << 3, cofs = lane & 15;
#pragma unroll
  for (int nj = 0; nj < 2; ++nj)
#pragma unroll
    for (int v = 0; v < 8; ++v) {
      int b  = mi * 16 + v + rofs;
      int jj = jb * 64 + nh * 32 + nj * 16 + cofs;
      const float* gxe = gx + (size_t)(t * Bb + b) * G3;
      float r = 1.f / (1.f + __expf(-(gxe[jj]           + acc[0][nj][v] + bhh[jj])));
      float z = 1.f / (1.f + __expf(-(gxe[NH + jj]      + acc[1][nj][v] + bhh[NH + jj])));
      float n = tanhf(gxe[2 * NH + jj] + r * (acc[2][nj][v] + bhh[2 * NH + jj]));
      float hp = hf_in[(size_t)b * NH + jj];
      float hn = (1.f - z) * n + z * hp;
      hf_out[(size_t)b * NH + jj] = hn;
      hb_out[(size_t)b * NH + jj] = f2bf(hn);
    }
}

// ---------------------------------------------------------------------------
// FC: out[64,768] = h_last @ W_fc^T + b_fc
// ---------------------------------------------------------------------------
__global__ void __launch_bounds__(256) k_fc(
    const unsigned short* __restrict__ hb, const unsigned short* __restrict__ wfc,
    const float* __restrict__ bfc, float* __restrict__ out)
{
  __shared__ __align__(16) unsigned short As[2][64 * KSTR];
  __shared__ __align__(16) unsigned short Bs[2][64 * KSTR];
  const int tid = threadIdx.x, lane = tid & 31, w = tid >> 5;
  const int jb = blockIdx.x;               // 12 blocks x 64 cols
  const int mi = w & 3, nh = w >> 2;
  const unsigned short* gB = wfc + (size_t)jb * 64 * NH;
  const unsigned ldsA = lds_off(&As[0][0]);
  const unsigned ldsB = lds_off(&Bs[0][0]);
  constexpr unsigned BUFB = 64 * KSTR * 2;
  v8f acc[2] = {};

  auto issue = [&](int buf, int k0) {
    int r = tid >> 2, c = (tid & 3) << 3;
    unsigned lo = (unsigned)buf * BUFB + (unsigned)(r * KSTR + c) * 2u;
    unsigned go = (unsigned)(r * NH + k0 + c) * 2u;
    async_copy_b128(ldsA + lo, hb, go);
    async_copy_b128(ldsB + lo, gB, go);
  };

  issue(0, 0);
  for (int k0 = 0; k0 < NH; k0 += 32) {
    const int cur = (k0 >> 5) & 1;
    if (k0 + 32 < NH) {
      issue(cur ^ 1, k0 + 32);
      asm volatile("s_wait_asynccnt 2" ::: "memory");
    } else {
      asm volatile("s_wait_asynccnt 0" ::: "memory");
    }
    __syncthreads();
    Frag a = load_fragA(As[cur], mi * 16, lane);
#pragma unroll
    for (int nj = 0; nj < 2; ++nj) {
      Frag bf = load_fragB(Bs[cur], nh * 32 + nj * 16, lane);
      acc[nj] = wmma_bf16(a, bf, acc[nj]);
    }
    __syncthreads();
  }

  const int rofs = (lane >> 4) << 3, cofs = lane & 15;
#pragma unroll
  for (int nj = 0; nj < 2; ++nj)
#pragma unroll
    for (int v = 0; v < 8; ++v) {
      int b = mi * 16 + v + rofs;
      int c = jb * 64 + nh * 32 + nj * 16 + cofs;
      out[(size_t)b * OUTC + c] = acc[nj][v] + bfc[c];
    }
}

// ---------------------------------------------------------------------------
extern "C" void kernel_launch(void* const* d_in, const int* in_sizes, int n_in,
                              void* d_out, int out_size, void* d_ws, size_t ws_size,
                              hipStream_t stream) {
  const float*     x   = (const float*)d_in[0];
  const long long* ei  = (const long long*)d_in[1];
  const float*     Wg  = (const float*)d_in[2];
  const float*     bg  = (const float*)d_in[3];
  const float*     Wih = (const float*)d_in[4];
  const float*     Whh = (const float*)d_in[5];
  const float*     bih = (const float*)d_in[6];
  const float*     bhh = (const float*)d_in[7];
  const float*     Wfc = (const float*)d_in[8];
  const float*     bfc = (const float*)d_in[9];
  float*           out = (float*)d_out;

  // Workspace carve-out (~165 MB): bf16 weights, bf16 h_seq, f32 gates_x, h ping-pong.
  char* p = (char*)d_ws;
  float*          dinv  = (float*)p;          p += 256;
  unsigned short* wih_b = (unsigned short*)p; p += (size_t)G3 * NH * 2;
  unsigned short* whh_b = (unsigned short*)p; p += (size_t)G3 * NH * 2;
  unsigned short* wfc_b = (unsigned short*)p; p += (size_t)OUTC * NH * 2;
  unsigned short* hseq  = (unsigned short*)p; p += (size_t)Ll * Bb * NH * 2;
  float*          gx    = (float*)p;          p += (size_t)Ll * Bb * G3 * 4;
  float*          hf    = (float*)p;          p += (size_t)2 * Bb * NH * 4;
  unsigned short* hb    = (unsigned short*)p; p += (size_t)2 * Bb * NH * 2;

  k_dinv<<<1, 64, 0, stream>>>(ei, dinv);
  k_f2bf<<<2048, 256, 0, stream>>>(Wih, wih_b, G3 * NH);
  k_f2bf<<<2048, 256, 0, stream>>>(Whh, whh_b, G3 * NH);
  k_f2bf<<<256, 256, 0, stream>>>(Wfc, wfc_b, OUTC * NH);
  k_zero_h<<<256, 256, 0, stream>>>(hf, hb, 2 * Bb * NH);

  k_gcn<<<dim3(Bb, Ll), 256, 0, stream>>>(x, ei, Wg, bg, dinv, hseq);
  k_gatesx<<<dim3(G3 / 128, (Ll * Bb) / 128), 256, 0, stream>>>(hseq, wih_b, bih, gx);

  for (int t = 0; t < Ll; ++t) {
    const unsigned short* hbi = hb + (size_t)(t & 1) * Bb * NH;
    const float*          hfi = hf + (size_t)(t & 1) * Bb * NH;
    unsigned short*       hbo = hb + (size_t)((t + 1) & 1) * Bb * NH;
    float*                hfo = hf + (size_t)((t + 1) & 1) * Bb * NH;
    k_gru_step<<<NH / 64, 256, 0, stream>>>(t, hbi, hfi, whh_b, bhh, gx, hfo, hbo);
  }

  // After 64 steps the live buffer index is (64 & 1) == 0.
  k_fc<<<OUTC / 64, 256, 0, stream>>>(hb, wfc_b, bfc, out);

  (void)in_sizes; (void)n_in; (void)out_size; (void)ws_size;
}
